// CausalSelfAttention_13400297964035
// MI455X (gfx1250) — matrix operand order repaired and checked by reading, not compiled, not run on hardware
//
#include <hip/hip_runtime.h>
#include <hip/hip_bf16.h>
#include <math.h>

typedef __attribute__((ext_vector_type(16))) _Float16 v16h;
typedef __attribute__((ext_vector_type(8)))  float    v8f;
typedef __attribute__((address_space(3))) _Float16 lds_f16;

constexpr int Bn = 4;
constexpr int Sn = 2048;
constexpr int Cn = 1024;
constexpr int Hn = 16;
constexpr int Dn = 64;
constexpr int Mn = Bn * Sn;   // 8192 total rows

// GEMM LDS weight-tile staging geometry
constexpr int WCOLS = 128;            // output cols per block
constexpr int WK    = 32;             // K depth per tile
constexpr int WLD   = 40;             // padded col stride (f16) -> bank-conflict-free
constexpr int WBUFE = WCOLS * WLD;    // elements per buffer
constexpr int WBUFB = WBUFE * 2;      // bytes per buffer

// ---------------- WMMA helper (f16 in, f32 accum) ----------------
__device__ __forceinline__ v8f wmma16(v16h a, v16h b, v8f c) {
  // 8 args: (neg_a, A, neg_b, B, c_mod, C, reuse_a, reuse_b)
  return __builtin_amdgcn_wmma_f32_16x16x32_f16(false, a, false, b, (short)0, c,
                                                false, false);
}

// A-matrix 16x32 f16 fragment. lane<16: row=lane, K {k0..k0+7, k0+16..k0+23};
// lane>=16: same row, K {k0+8..15, k0+24..31}. base points at (row0, k0).
__device__ __forceinline__ v16h load_a_f16(const _Float16* base, int ld, int lane) {
  int r = lane & 15, hf = lane >> 4;
  const _Float16* p0 = base + (size_t)r * ld + hf * 8;
  const _Float16* p1 = p0 + 16;
  v16h a;
#pragma unroll
  for (int i = 0; i < 8; ++i) { a[i] = p0[i]; a[i + 8] = p1[i]; }
  return a;
}

// B-matrix 32x16 f16 fragment. lane<16: col=lane, K {k0..k0+7, k0+16..k0+23};
// lane>=16: same col, K {k0+8..15, k0+24..31}. Memory holds columns contiguously in K
// (stride ld between columns); base points at (k0, col0).
__device__ __forceinline__ v16h load_b_f16(const _Float16* base, int ld, int lane) {
  int r = lane & 15, hf = lane >> 4;
  const _Float16* p = base + (size_t)r * ld + hf * 8;
  v16h b;
#pragma unroll
  for (int i = 0; i < 8; ++i) { b[i] = p[i]; b[i + 8] = p[i + 16]; }
  return b;
}

// Issue one 16-byte async global->LDS copy (ASYNCcnt-tracked).
__device__ __forceinline__ void async_copy16(unsigned lds_off, const _Float16* g) {
  unsigned long long ga = (unsigned long long)(uintptr_t)g;
  asm volatile("global_load_async_to_lds_b128 %0, %1, off"
               :: "v"(lds_off), "v"(ga) : "memory");
}

// ---------------- Kernel 0: one-shot fp32 -> f16 conversion ----------------
__global__ __launch_bounds__(256) void cvt_f16_kernel(
    const float* __restrict__ src, _Float16* __restrict__ dst, int n8) {
  int idx = blockIdx.x * 256 + threadIdx.x;
  if (idx >= n8) return;
  size_t base = (size_t)idx * 8;
#pragma unroll
  for (int i = 0; i < 8; ++i) dst[base + i] = (_Float16)src[base + i];
}

// ---------------- Kernel 1: fused QKV projection ----------------
// out = X @ W^T + b with X,W f16. Weight K-slices are staged into LDS with
// double-buffered async copies (shared by all 8 waves); WMMA-B frags come from
// ds_load_b128 and are register-double-buffered so WMMAs don't stall on DScnt.
// A fragments are software-pipelined across k-iterations.
// z = 0:Q (scaled 1/8) 1:K 2:V (transposed [B,H,d,S]).
__global__ __launch_bounds__(256) void qkv_proj_kernel(
    const _Float16* __restrict__ X,
    const _Float16* __restrict__ Wqh, const float* __restrict__ bq,
    const _Float16* __restrict__ Wkh, const float* __restrict__ bk,
    const _Float16* __restrict__ Wvh, const float* __restrict__ bv,
    _Float16* __restrict__ Qo, _Float16* __restrict__ Ko,
    _Float16* __restrict__ Vt) {
  __shared__ _Float16 Wt[2 * WBUFE];

  int lane = threadIdx.x & 31, w = threadIdx.x >> 5;
  int r = lane & 15, hf = lane >> 4;
  int z = blockIdx.z;
  const _Float16* W    = (z == 0) ? Wqh : (z == 1) ? Wkh : Wvh;
  const float*    bias = (z == 0) ? bq  : (z == 1) ? bk  : bv;

  int m0 = blockIdx.y * 128 + w * 16;   // 16 rows per wave
  int n0 = blockIdx.x * WCOLS;          // 128 output cols per block

  // Per-thread chunk assignment: 512 x 16B chunks, 2 per thread.
  // chunk c: col = c>>2, kpart = (c&3)*8 (f16 elems)
  int c0 = threadIdx.x, c1 = threadIdx.x + 256;
  unsigned ldsbase = (unsigned)(uintptr_t)(lds_f16*)&Wt[0];
  unsigned l0 = ldsbase + (unsigned)(((c0 >> 2) * WLD + (c0 & 3) * 8) * 2);
  unsigned l1 = ldsbase + (unsigned)(((c1 >> 2) * WLD + (c1 & 3) * 8) * 2);
  const _Float16* g0 = W + (size_t)(n0 + (c0 >> 2)) * Cn + (c0 & 3) * 8;
  const _Float16* g1 = W + (size_t)(n0 + (c1 >> 2)) * Cn + (c1 & 3) * 8;

  v8f acc[8];
#pragma unroll
  for (int t = 0; t < 8; ++t) {
    float bv0 = bias[n0 + t * 16 + r];
#pragma unroll
    for (int i = 0; i < 8; ++i) acc[t][i] = bv0;
  }

  // Preload tile 0 into buffer 0
  async_copy16(l0, g0);
  async_copy16(l1, g1);

  const _Float16* Abase = X + (size_t)m0 * Cn;
  v16h a_cur = load_a_f16(Abase, Cn, lane);   // k0 = 0 A fragment

  for (int k0 = 0; k0 < Cn; k0 += WK) {
    int cur = (k0 >> 5) & 1;
    bool more = (k0 + WK) < Cn;   // block-uniform
    if (more) {   // issue next tile into the other buffer
      unsigned bo = (cur ^ 1) * WBUFB;
      async_copy16(l0 + bo, g0 + k0 + WK);
      async_copy16(l1 + bo, g1 + k0 + WK);
      asm volatile("s_wait_asynccnt 2" ::: "memory");  // current tile done
    } else {
      asm volatile("s_wait_asynccnt 0" ::: "memory");
    }
    __syncthreads();      // tile fully visible to all waves

    // Pipeline next A fragment under this tile's 8 WMMAs
    v16h a_next;
    if (more) a_next = load_a_f16(Abase + k0 + WK, Cn, lane);

    const _Float16* Wl = &Wt[cur * WBUFE];
    // Register-double-buffered B fragments from LDS
    v16h bf = load_b_f16(Wl, WLD, lane);
#pragma unroll
    for (int t = 0; t < 8; ++t) {
      v16h bn = (t < 7) ? load_b_f16(Wl + (size_t)((t + 1) * 16) * WLD, WLD, lane)
                        : bf;
      acc[t] = wmma16(a_cur, bf, acc[t]);
      bf = bn;
    }
    if (more) a_cur = a_next;
    __syncthreads();      // all waves done reading 'cur' before it is refilled
  }

#pragma unroll
  for (int t = 0; t < 8; ++t) {
    int cg = n0 + t * 16 + r;          // global output column
    int h = cg >> 6, dd = cg & 63;     // head, dim-in-head
#pragma unroll
    for (int i = 0; i < 8; ++i) {
      int mg = m0 + i + 8 * hf;        // global row
      int b = mg >> 11, s = mg & (Sn - 1);
      size_t bh = (size_t)b * Hn + h;
      if (z == 0)
        Qo[(bh * Sn + s) * Dn + dd] = (_Float16)(acc[t][i] * 0.125f);
      else if (z == 1)
        Ko[(bh * Sn + s) * Dn + dd] = (_Float16)acc[t][i];
      else
        Vt[(bh * Dn + dd) * Sn + s] = (_Float16)acc[t][i];
    }
  }
}

// ---------------- Kernel 2: causal flash attention ----------------
// One wave = 16 query rows; 8 waves/block cover a 128-row query tile for one
// (b,h). Streams 32-key blocks with online softmax; P re-laid-out C->A via LDS.
// V fragments are issued at the top of each iteration (independent of scores)
// so their latency hides under the score WMMAs + softmax.
__global__ __launch_bounds__(256) void attn_kernel(
    const _Float16* __restrict__ Q, const _Float16* __restrict__ K,
    const _Float16* __restrict__ Vt, const float* __restrict__ pad,
    _Float16* __restrict__ Y) {
  __shared__ _Float16 Pb[8][16 * 32];  // per-wave P tile (16 rows x 32 keys)

  int lane = threadIdx.x & 31, w = threadIdx.x >> 5;
  int r = lane & 15, hf = lane >> 4;
  int b = blockIdx.z, h = blockIdx.y;
  int m0 = blockIdx.x * 128 + w * 16;

  size_t bh = (size_t)b * Hn + h;
  const _Float16* Qh = Q + bh * Sn * Dn;
  const _Float16* Kh = K + bh * Sn * Dn;
  const _Float16* Vh = Vt + bh * Dn * Sn;
  const float* padb = pad + (size_t)b * Sn * Sn;
  _Float16* myP = &Pb[w][0];

  // Hoist Q fragments (16 rows x 64 dims = two K=32 A-frags)
  v16h aQ0 = load_a_f16(Qh + (size_t)m0 * Dn, Dn, lane);
  v16h aQ1 = load_a_f16(Qh + (size_t)m0 * Dn + 32, Dn, lane);

  float mi[8], li[8];
  v8f o[4];
#pragma unroll
  for (int i = 0; i < 8; ++i) { mi[i] = -__builtin_inff(); li[i] = 0.f; }
#pragma unroll
  for (int t = 0; t < 4; ++t)
#pragma unroll
    for (int i = 0; i < 8; ++i) o[t][i] = 0.f;

  for (int kb = 0; kb < m0 + 16; kb += 32) {   // causal bound (wave-uniform)
    // Early-issue V fragments: independent of the score path, so these global
    // loads overlap the 4 score WMMAs + the whole softmax.
    v16h bv[4];
#pragma unroll
    for (int t = 0; t < 4; ++t)
      bv[t] = load_b_f16(Vh + (size_t)(t * 16) * Sn + kb, Sn, lane);

    // All 4 K fragments up front -> one wait, 4 back-to-back WMMAs
    v16h bk0 = load_b_f16(Kh + (size_t)kb * Dn, Dn, lane);
    v16h bk1 = load_b_f16(Kh + (size_t)kb * Dn + 32, Dn, lane);
    v16h bk2 = load_b_f16(Kh + (size_t)(kb + 16) * Dn, Dn, lane);
    v16h bk3 = load_b_f16(Kh + (size_t)(kb + 16) * Dn + 32, Dn, lane);

    // Scores: C initialized with additive pad mask (free add via accumulator)
    v8f s0, s1;
#pragma unroll
    for (int i = 0; i < 8; ++i) {
      int q = m0 + i + 8 * hf;
      s0[i] = padb[(size_t)q * Sn + kb + r];
      s1[i] = padb[(size_t)q * Sn + kb + 16 + r];
    }
    s0 = wmma16(aQ0, bk0, s0);
    s0 = wmma16(aQ1, bk1, s0);
    s1 = wmma16(aQ0, bk2, s1);
    s1 = wmma16(aQ1, bk3, s1);

    // Causal mask (VALU selects, no divergence)
#pragma unroll
    for (int i = 0; i < 8; ++i) {
      int q = m0 + i + 8 * hf;
      if (kb + r > q)      s0[i] = -__builtin_inff();
      if (kb + 16 + r > q) s1[i] = -__builtin_inff();
    }

    // Online softmax; rows live in 16-lane halves -> xor-shuffles 1,2,4,8
    float alpha[8];
#pragma unroll
    for (int i = 0; i < 8; ++i) {
      float t0 = fmaxf(s0[i], s1[i]);
      t0 = fmaxf(t0, __shfl_xor(t0, 1, 32));
      t0 = fmaxf(t0, __shfl_xor(t0, 2, 32));
      t0 = fmaxf(t0, __shfl_xor(t0, 4, 32));
      t0 = fmaxf(t0, __shfl_xor(t0, 8, 32));
      float mn = fmaxf(mi[i], t0);
      alpha[i] = __expf(mi[i] - mn);
      mi[i] = mn;
      float p0 = __expf(s0[i] - mn);
      float p1 = __expf(s1[i] - mn);
      int prow = i + 8 * hf;
      myP[prow * 32 + r]      = (_Float16)p0;
      myP[prow * 32 + 16 + r] = (_Float16)p1;
      float rs = p0 + p1;
      rs += __shfl_xor(rs, 1, 32);
      rs += __shfl_xor(rs, 2, 32);
      rs += __shfl_xor(rs, 4, 32);
      rs += __shfl_xor(rs, 8, 32);
      li[i] = li[i] * alpha[i] + rs;
    }
#pragma unroll
    for (int t = 0; t < 4; ++t)
#pragma unroll
      for (int i = 0; i < 8; ++i) o[t][i] *= alpha[i];

    // O += P @ V ; P reloaded from LDS in A layout (DS ops in-order per wave);
    // V fragments were issued at iteration start and are already resident.
    v16h aP = load_a_f16(myP, 32, lane);
#pragma unroll
    for (int t = 0; t < 4; ++t)
      o[t] = wmma16(aP, bv[t], o[t]);
  }

#pragma unroll
  for (int i = 0; i < 8; ++i) li[i] = 1.0f / li[i];
#pragma unroll
  for (int t = 0; t < 4; ++t) {
    int cg = h * Dn + t * 16 + r;      // column in [B,S,C] layout
#pragma unroll
    for (int i = 0; i < 8; ++i) {
      int s = m0 + i + 8 * hf;
      Y[((size_t)b * Sn + s) * Cn + cg] = (_Float16)(o[t][i] * li[i]);
    }
  }
}

// ---------------- Kernel 3: output projection ----------------
__global__ __launch_bounds__(256) void out_proj_kernel(
    const _Float16* __restrict__ Y, const _Float16* __restrict__ Wph,
    const float* __restrict__ bp, float* __restrict__ out) {
  __shared__ _Float16 Wt[2 * WBUFE];

  int lane = threadIdx.x & 31, w = threadIdx.x >> 5;
  int r = lane & 15, hf = lane >> 4;
  int m0 = blockIdx.y * 128 + w * 16;
  int n0 = blockIdx.x * WCOLS;

  int c0 = threadIdx.x, c1 = threadIdx.x + 256;
  unsigned ldsbase = (unsigned)(uintptr_t)(lds_f16*)&Wt[0];
  unsigned l0 = ldsbase + (unsigned)(((c0 >> 2) * WLD + (c0 & 3) * 8) * 2);
  unsigned l1 = ldsbase + (unsigned)(((c1 >> 2) * WLD + (c1 & 3) * 8) * 2);
  const _Float16* g0 = Wph + (size_t)(n0 + (c0 >> 2)) * Cn + (c0 & 3) * 8;
  const _Float16* g1 = Wph + (size_t)(n0 + ((c1) >> 2)) * Cn + (c1 & 3) * 8;

  v8f acc[8];
#pragma unroll
  for (int t = 0; t < 8; ++t) {
    float bv0 = bp[n0 + t * 16 + r];
#pragma unroll
    for (int i = 0; i < 8; ++i) acc[t][i] = bv0;
  }

  async_copy16(l0, g0);
  async_copy16(l1, g1);

  const _Float16* Abase = Y + (size_t)m0 * Cn;
  v16h a_cur = load_a_f16(Abase, Cn, lane);

  for (int k0 = 0; k0 < Cn; k0 += WK) {
    int cur = (k0 >> 5) & 1;
    bool more = (k0 + WK) < Cn;
    if (more) {
      unsigned bo = (cur ^ 1) * WBUFB;
      async_copy16(l0 + bo, g0 + k0 + WK);
      async_copy16(l1 + bo, g1 + k0 + WK);
      asm volatile("s_wait_asynccnt 2" ::: "memory");
    } else {
      asm volatile("s_wait_asynccnt 0" ::: "memory");
    }
    __syncthreads();

    v16h a_next;
    if (more) a_next = load_a_f16(Abase + k0 + WK, Cn, lane);

    const _Float16* Wl = &Wt[cur * WBUFE];
    v16h bf = load_b_f16(Wl, WLD, lane);
#pragma unroll
    for (int t = 0; t < 8; ++t) {
      v16h bn = (t < 7) ? load_b_f16(Wl + (size_t)((t + 1) * 16) * WLD, WLD, lane)
                        : bf;
      acc[t] = wmma16(a_cur, bf, acc[t]);
      bf = bn;
    }
    if (more) a_cur = a_next;
    __syncthreads();
  }

#pragma unroll
  for (int t = 0; t < 8; ++t) {
    int cg = n0 + t * 16 + r;
#pragma unroll
    for (int i = 0; i < 8; ++i) {
      int mg = m0 + i + 8 * hf;
      out[(size_t)mg * Cn + cg] = acc[t][i];
    }
  }
}

extern "C" void kernel_launch(void* const* d_in, const int* in_sizes, int n_in,
                              void* d_out, int out_size, void* d_ws,
                              size_t ws_size, hipStream_t stream) {
  const float* tgt = (const float*)d_in[0];
  const float* pad = (const float*)d_in[1];
  const float* Wq  = (const float*)d_in[2];
  const float* bq  = (const float*)d_in[3];
  const float* Wk  = (const float*)d_in[4];
  const float* bk  = (const float*)d_in[5];
  const float* Wv  = (const float*)d_in[6];
  const float* bv  = (const float*)d_in[7];
  const float* Wp  = (const float*)d_in[8];
  const float* bp  = (const float*)d_in[9];
  float* out = (float*)d_out;

  // Workspace layout (f16):
  //  Q 16MB | K 16MB | V^T 16MB | Y 16MB | Xh 16MB | Wq/Wk/Wv/Wp 2MB each
  constexpr size_t MB = 1024 * 1024;
  char* ws = (char*)d_ws;
  _Float16* Q   = (_Float16*)(ws);
  _Float16* K   = (_Float16*)(ws + 16 * MB);
  _Float16* Vt  = (_Float16*)(ws + 32 * MB);
  _Float16* Y   = (_Float16*)(ws + 48 * MB);
  _Float16* Xh  = (_Float16*)(ws + 64 * MB);
  _Float16* Wqh = (_Float16*)(ws + 80 * MB);
  _Float16* Wkh = (_Float16*)(ws + 82 * MB);
  _Float16* Wvh = (_Float16*)(ws + 84 * MB);
  _Float16* Wph = (_Float16*)(ws + 86 * MB);

  dim3 blk(256);

  // Stage 0: fp32 -> f16 conversions (one pass each)
  int nX = Mn * Cn / 8;           // 8 elements/thread
  int nW = Cn * Cn / 8;
  cvt_f16_kernel<<<dim3((nX + 255) / 256), blk, 0, stream>>>(tgt, Xh, nX);
  cvt_f16_kernel<<<dim3((nW + 255) / 256), blk, 0, stream>>>(Wq, Wqh, nW);
  cvt_f16_kernel<<<dim3((nW + 255) / 256), blk, 0, stream>>>(Wk, Wkh, nW);
  cvt_f16_kernel<<<dim3((nW + 255) / 256), blk, 0, stream>>>(Wv, Wvh, nW);
  cvt_f16_kernel<<<dim3((nW + 255) / 256), blk, 0, stream>>>(Wp, Wph, nW);

  // Stage 1: QKV projections
  dim3 g1(Cn / WCOLS, Mn / 128, 3);
  qkv_proj_kernel<<<g1, blk, 0, stream>>>(Xh, Wqh, bq, Wkh, bk, Wvh, bv,
                                          Q, K, Vt);

  // Stage 2: causal flash attention
  dim3 g2(Sn / 128, Hn, Bn);
  attn_kernel<<<g2, blk, 0, stream>>>(Q, K, Vt, pad, Y);

  // Stage 3: output projection
  dim3 g3(Cn / WCOLS, Mn / 128, 1);
  out_proj_kernel<<<g3, blk, 0, stream>>>(Y, Wph, bp, out);
}
